// MSA_32341103739168
// MI455X (gfx1250) — matrix operand order repaired and checked by reading, not compiled, not run on hardware
//
#include <hip/hip_runtime.h>
#include <hip/hip_bf16.h>

// ---------------------------------------------------------------------------
// Multi-head self-attention for gfx1250 (MI455X), f16 WMMA with f32 accum.
// B=4, N=2048, DIM=1024, H=16, D=64.
// ---------------------------------------------------------------------------

#define BATCH     4
#define N_SEQ     2048
#define DIM_C     1024
#define NUM_HEADS 16
#define HEAD_DIM  64
#define SCALE_F   0.125f          // HEAD_DIM^-0.5
#define BN        (BATCH * N_SEQ) // 8192

typedef _Float16 v8h  __attribute__((ext_vector_type(8)));
typedef _Float16 v16h __attribute__((ext_vector_type(16)));
typedef float    v8f  __attribute__((ext_vector_type(8)));
typedef int      v4i  __attribute__((ext_vector_type(4)));

// CDNA5 async global->LDS (ASYNCcnt-tracked), guarded with fallback.
#if defined(__has_builtin)
#if __has_builtin(__builtin_amdgcn_global_load_async_to_lds_b128)
#define HAVE_ASYNC_LDS 1
#endif
#endif
#ifndef HAVE_ASYNC_LDS
#define HAVE_ASYNC_LDS 0
#endif

// ---------------------------------------------------------------------------
// Fragment loaders (wave32), contiguous k order used by this file (not the ISA table order).
//
// A (16x32 f16, M x K): lane L<16 -> row M=L, K = k0..k0+15 (contiguous)
//                       lane L>=16 -> row M=L-16, K = k0+16..k0+31
// B (32x16 f16, K x N): lane L<16 -> col N=L,    K = k0..k0+15 (contiguous)
//                       lane L>=16 -> col N=L-16, K = k0+16..k0+31
// C/D (16x16 f32):      lane L<16 -> col N=L, VGPR v -> row M=v
//                       lane L>=16 -> col N=L-16, VGPR v -> row M=8+v
// ---------------------------------------------------------------------------

__device__ __forceinline__ v16h load_a16(const _Float16* __restrict__ base,
                                         int ld, int k0, int lane) {
  const int m  = lane & 15;
  const int ko = (lane >> 4) << 4;                 // 0 or 16
  const _Float16* p = base + (size_t)m * ld + k0 + ko;
  v8h lo = *(const v8h*)(p);
  v8h hi = *(const v8h*)(p + 8);
  v16h r;
#pragma unroll
  for (int i = 0; i < 8; ++i) { r[i] = lo[i]; r[8 + i] = hi[i]; }
  return r;
}

// B[k][n] where column n of B is stored contiguously: colbase + n*ld + k
__device__ __forceinline__ v16h load_b16(const _Float16* __restrict__ colbase,
                                         int ld, int k0, int lane) {
  const int n  = lane & 15;
  const int ko = (lane >> 4) << 4;                 // 0 or 16
  return *(const v16h*)(colbase + (size_t)n * ld + k0 + ko);
}

// Same B-fragment load but from an LDS-staged tile [64 cols][128 k] halfs.
__device__ __forceinline__ v16h load_b16_lds(const _Float16* buf, int t,
                                             int ks, int lane) {
  const int n  = lane & 15;
  const int ko = (lane >> 4) << 4;
  return *(const v16h*)(buf + (size_t)(t * 16 + n) * 128 + ks + ko);
}

__device__ __forceinline__ v8f wmma_f16(v16h a, v16h b, v8f c) {
  return __builtin_amdgcn_wmma_f32_16x16x32_f16(false, a, false, b,
                                                (short)0, c, false, false);
}

// 16-byte global -> LDS copy (async path when available).
// Builtin signature: (AS1 v4i*, AS3 v4i*, Imm offset, Imm cpol).
__device__ __forceinline__ void copy16_g2l(const _Float16* g, _Float16* l) {
#if HAVE_ASYNC_LDS
  v4i* gg = (v4i*)g;
  v4i* ll = (v4i*)l;
  __builtin_amdgcn_global_load_async_to_lds_b128(
      (__attribute__((address_space(1))) v4i*)gg,
      (__attribute__((address_space(3))) v4i*)ll, 0, 0);
#else
  *(v8h*)l = *(const v8h*)g;
#endif
}

__device__ __forceinline__ void wait_async_lds() {
#if HAVE_ASYNC_LDS
  asm volatile("s_wait_asynccnt 0x0" ::: "memory");
#endif
}

// ---------------------------------------------------------------------------
// 1) f32 -> f16 conversion
// ---------------------------------------------------------------------------
__global__ void cvt_f2h(const float* __restrict__ src,
                        _Float16* __restrict__ dst, int n) {
  int i = blockIdx.x * blockDim.x + threadIdx.x;
  const int stride = gridDim.x * blockDim.x;
  for (; i < n; i += stride) dst[i] = (_Float16)src[i];
}

// ---------------------------------------------------------------------------
// 2) Pack int32 mask [B,N,N] into bitmask [B,N,N/32] (bit=1 means keep).
// ---------------------------------------------------------------------------
__global__ void pack_mask(const int* __restrict__ mask,
                          unsigned int* __restrict__ pm, int total) {
  int i = blockIdx.x * blockDim.x + threadIdx.x;
  const int stride = gridDim.x * blockDim.x;
  const int NW = N_SEQ / 32;
  for (; i < total; i += stride) {
    const int row = i / NW, w = i % NW;
    const int* p = mask + (size_t)row * N_SEQ + w * 32;
    unsigned int bits = 0;
#pragma unroll
    for (int j = 0; j < 32; ++j) bits |= (p[j] != 0 ? 1u : 0u) << j;
    pm[i] = bits;
  }
}

// ---------------------------------------------------------------------------
// Shared GEMM core: 4-wave workgroup computes 64 rows x 64 cols of
// Y = A @ W^T. The W tile (64 cols x 128 k) is staged into LDS via async
// global->LDS and shared by all 4 waves. A fragments are software-pipelined
// (next k-step preloaded before consuming current); the 4 LDS B fragments
// are batch-loaded before the 4 WMMAs so dscnt waits can be partial.
// ---------------------------------------------------------------------------
__device__ __forceinline__ void gemm_core(const _Float16* __restrict__ Arow,
                                          const _Float16* __restrict__ W,
                                          int c0, int tid, int lane,
                                          _Float16* ldsB, v8f acc[4]) {
  v16h a = load_a16(Arow, DIM_C, 0, lane);       // prologue A fragment
  for (int kb = 0; kb < DIM_C; kb += 128) {
    __syncthreads();                             // previous tile consumed
#pragma unroll
    for (int i = 0; i < 8; ++i) {                // 1024 x 16B chunks / 128 thr
      const int chunk = tid + i * 128;
      const int col = chunk >> 4, kc = chunk & 15;
      copy16_g2l(W + (size_t)(c0 + col) * DIM_C + kb + kc * 8,
                 ldsB + (size_t)col * 128 + kc * 8);
    }
    wait_async_lds();
    __syncthreads();                             // staged tile visible
#pragma unroll
    for (int ks = 0; ks < 128; ks += 32) {
      v16h b0 = load_b16_lds(ldsB, 0, ks, lane);
      v16h b1 = load_b16_lds(ldsB, 1, ks, lane);
      v16h b2 = load_b16_lds(ldsB, 2, ks, lane);
      v16h b3 = load_b16_lds(ldsB, 3, ks, lane);
      const int kn = kb + ks + 32;               // preload next A fragment
      v16h an = load_a16(Arow, DIM_C, (kn < DIM_C) ? kn : 0, lane);
      acc[0] = wmma_f16(a, b0, acc[0]);
      acc[1] = wmma_f16(a, b1, acc[1]);
      acc[2] = wmma_f16(a, b2, acc[2]);
      acc[3] = wmma_f16(a, b3, acc[3]);
      a = an;
    }
  }
}

// ---------------------------------------------------------------------------
// 3) QKV projection: y = x @ W^T + bias.
//    sel 0 -> Q [B,H,N,D], sel 1 -> K [B,H,N,D], sel 2 -> V^T [B,H,D,N].
// ---------------------------------------------------------------------------
__global__ void __launch_bounds__(128)
gemm_qkv(const _Float16* __restrict__ xh,
         const _Float16* __restrict__ Wqh,
         const _Float16* __restrict__ Wkh,
         const _Float16* __restrict__ Wvh,
         const float* __restrict__ bq,
         const float* __restrict__ bk,
         const float* __restrict__ bv,
         _Float16* __restrict__ Qh,
         _Float16* __restrict__ Kh,
         _Float16* __restrict__ Vth) {
  __shared__ _Float16 ldsB[64 * 128];
  const int tid  = threadIdx.x;
  const int lane = tid & 31, wave = tid >> 5;
  const int r0   = blockIdx.x * 64 + wave * 16;
  const int c0   = blockIdx.y * 64;
  const int sel  = blockIdx.z;

  const _Float16* W    = (sel == 0) ? Wqh : (sel == 1) ? Wkh : Wvh;
  const float*    bias = (sel == 0) ? bq  : (sel == 1) ? bk  : bv;

  v8f acc[4] = {};
  gemm_core(xh + (size_t)r0 * DIM_C, W, c0, tid, lane, ldsB, acc);

  const int col  = lane & 15;
  const int roff = (lane >> 4) << 3;
#pragma unroll
  for (int t = 0; t < 4; ++t) {
    const int c = c0 + t * 16 + col;
    const float bval = bias[c];
    const int h = c >> 6, d = c & 63;
#pragma unroll
    for (int v = 0; v < 8; ++v) {
      const int bn = r0 + roff + v;
      const int b = bn >> 11, n = bn & (N_SEQ - 1);
      const _Float16 hv = (_Float16)(acc[t][v] + bval);
      if (sel == 0)
        Qh[(((size_t)(b * NUM_HEADS + h)) * N_SEQ + n) * HEAD_DIM + d] = hv;
      else if (sel == 1)
        Kh[(((size_t)(b * NUM_HEADS + h)) * N_SEQ + n) * HEAD_DIM + d] = hv;
      else
        Vth[(((size_t)(b * NUM_HEADS + h)) * HEAD_DIM + d) * N_SEQ + n] = hv;
    }
  }
}

// ---------------------------------------------------------------------------
// 4) Flash attention, one wave per (b, h, 32-query tile).
//    Computes S^T = K · Q^T so the C-layout column (= lane) is a query row:
//    softmax state (M, L, rescale) is per-lane scalar; shfl_xor(16) merges
//    half-wave row groups and converts P into the B-fragment for P·V.
//    Two query tiles per wave share the K / V^T A-fragments.
// ---------------------------------------------------------------------------
__global__ void __launch_bounds__(32)
attention(const _Float16* __restrict__ Qh,
          const _Float16* __restrict__ Kh,
          const _Float16* __restrict__ Vth,
          const unsigned int* __restrict__ pmask,
          _Float16* __restrict__ AOh) {
  const int lane = threadIdx.x;
  const int n0   = blockIdx.x * 32;
  const int h    = blockIdx.y;
  const int b    = blockIdx.z;
  const size_t hb = (size_t)b * NUM_HEADS + h;

  const _Float16* Qp = Qh + (hb * N_SEQ + n0) * HEAD_DIM;
  const _Float16* Kp = Kh + hb * N_SEQ * HEAD_DIM;
  const _Float16* Vp = Vth + hb * HEAD_DIM * N_SEQ;
  const unsigned int* pm[2];
  pm[0] = pmask + ((size_t)b * N_SEQ + (n0 + (lane & 15))) * (N_SEQ / 32);
  pm[1] = pm[0] + (size_t)16 * (N_SEQ / 32);

  // loop-invariant Q fragments (B operand of S^T = K·Q^T), per query tile
  v16h bq[2][2];
#pragma unroll
  for (int q = 0; q < 2; ++q) {
    bq[q][0] = load_b16(Qp + (size_t)q * 16 * HEAD_DIM, HEAD_DIM, 0, lane);
    bq[q][1] = load_b16(Qp + (size_t)q * 16 * HEAD_DIM, HEAD_DIM, 32, lane);
  }

  v8f o[2][4] = {};              // O^T accumulators: [qtile][d-tile]
  float M[2] = {-1e30f, -1e30f}, L[2] = {0.0f, 0.0f};
  const bool lo16 = (lane < 16);
  const int bitoff = (lane >> 4) << 3;

  for (int m0 = 0; m0 < N_SEQ; m0 += 32) {
    if (m0 + 32 < N_SEQ)         // near-cache prefetch of next K tile
      __builtin_prefetch(Kp + (size_t)(m0 + 32 + lane) * HEAD_DIM, 0, 3);

    // shared A-fragments: K rows (2 m-tiles x 2 d-halves), V^T rows (4 d-tiles)
    v16h ak[2][2], av[4];
#pragma unroll
    for (int mt = 0; mt < 2; ++mt) {
      ak[mt][0] = load_a16(Kp + (size_t)(m0 + mt * 16) * HEAD_DIM, HEAD_DIM, 0, lane);
      ak[mt][1] = load_a16(Kp + (size_t)(m0 + mt * 16) * HEAD_DIM, HEAD_DIM, 32, lane);
    }
#pragma unroll
    for (int t = 0; t < 4; ++t)
      av[t] = load_a16(Vp + (size_t)(t * 16) * N_SEQ + m0, N_SEQ, 0, lane);

#pragma unroll
    for (int q = 0; q < 2; ++q) {
      v8f sa = {}, sb = {};
      sa = wmma_f16(ak[0][0], bq[q][0], sa);
      sa = wmma_f16(ak[0][1], bq[q][1], sa);
      sb = wmma_f16(ak[1][0], bq[q][0], sb);
      sb = wmma_f16(ak[1][1], bq[q][1], sb);

      const unsigned int w = pm[q][m0 >> 5];

      float pa[8], pb[8];
      float tmax = -1e30f;
#pragma unroll
      for (int v = 0; v < 8; ++v) {
        float s0 = sa[v] * SCALE_F;
        float s1 = sb[v] * SCALE_F;
        s0 = ((w >> (bitoff + v)) & 1u)      ? s0 : -1e9f;
        s1 = ((w >> (16 + bitoff + v)) & 1u) ? s1 : -1e9f;
        pa[v] = s0; pb[v] = s1;
        tmax = fmaxf(tmax, fmaxf(s0, s1));
      }
      tmax = fmaxf(tmax, __shfl_xor(tmax, 16, 32));
      const float Mn = fmaxf(M[q], tmax);
      const float f  = __expf(M[q] - Mn);
      M[q] = Mn;

      float lsum = 0.0f;
#pragma unroll
      for (int v = 0; v < 8; ++v) {
        pa[v] = __expf(pa[v] - Mn);
        pb[v] = __expf(pb[v] - Mn);
        lsum += pa[v] + pb[v];
      }
      lsum += __shfl_xor(lsum, 16, 32);
      L[q] = L[q] * f + lsum;

#pragma unroll
      for (int t = 0; t < 4; ++t)
#pragma unroll
        for (int v = 0; v < 8; ++v) o[q][t][v] *= f;

      // P^T (32m x 16n) C-layout -> B-fragment (k=m, col=n) via xor-16 swaps
      v16h pf;
#pragma unroll
      for (int v = 0; v < 8; ++v) {
        const float oa = __shfl_xor(pa[v], 16, 32);
        const float ob = __shfl_xor(pb[v], 16, 32);
        pf[v]     = (_Float16)(lo16 ? pa[v] : ob);
        pf[8 + v] = (_Float16)(lo16 ? oa    : pb[v]);
      }

#pragma unroll
      for (int t = 0; t < 4; ++t) o[q][t] = wmma_f16(av[t], pf, o[q][t]);
    }
  }

  const int col  = lane & 15;
  const int roff = (lane >> 4) << 3;
#pragma unroll
  for (int q = 0; q < 2; ++q) {
    const float invL = 1.0f / L[q];
    const int n = n0 + q * 16 + col;
#pragma unroll
    for (int t = 0; t < 4; ++t)
#pragma unroll
      for (int v = 0; v < 8; ++v) {
        const int d = t * 16 + roff + v;
        AOh[((size_t)b * N_SEQ + n) * DIM_C + h * HEAD_DIM + d] =
            (_Float16)(o[q][t][v] * invL);
      }
  }
}

// ---------------------------------------------------------------------------
// 5) Output projection: out = AO @ Wo^T + bo, f32 output.
// ---------------------------------------------------------------------------
__global__ void __launch_bounds__(128)
gemm_out(const _Float16* __restrict__ AOh,
         const _Float16* __restrict__ Woh,
         const float* __restrict__ bo,
         float* __restrict__ out) {
  __shared__ _Float16 ldsB[64 * 128];
  const int tid  = threadIdx.x;
  const int lane = tid & 31, wave = tid >> 5;
  const int r0   = blockIdx.x * 64 + wave * 16;
  const int c0   = blockIdx.y * 64;

  v8f acc[4] = {};
  gemm_core(AOh + (size_t)r0 * DIM_C, Woh, c0, tid, lane, ldsB, acc);

  const int col  = lane & 15;
  const int roff = (lane >> 4) << 3;
#pragma unroll
  for (int t = 0; t < 4; ++t) {
    const int c = c0 + t * 16 + col;
    const float bval = bo[c];
#pragma unroll
    for (int v = 0; v < 8; ++v) {
      const int r = r0 + roff + v;
      out[(size_t)r * DIM_C + c] = acc[t][v] + bval;
    }
  }
}

// ---------------------------------------------------------------------------
// Launch
// ---------------------------------------------------------------------------
extern "C" void kernel_launch(void* const* d_in, const int* in_sizes, int n_in,
                              void* d_out, int out_size, void* d_ws, size_t ws_size,
                              hipStream_t stream) {
  const float* x    = (const float*)d_in[0];
  const int*   mask = (const int*)  d_in[1];
  const float* Wq   = (const float*)d_in[2];
  const float* bq   = (const float*)d_in[3];
  const float* Wk   = (const float*)d_in[4];
  const float* bk   = (const float*)d_in[5];
  const float* Wv   = (const float*)d_in[6];
  const float* bv   = (const float*)d_in[7];
  const float* Wo   = (const float*)d_in[8];
  const float* bo   = (const float*)d_in[9];
  float* out = (float*)d_out;

  char* ws = (char*)d_ws;
  const size_t SZ_X = (size_t)BN * DIM_C * 2;        // 16 MB
  const size_t SZ_W = (size_t)DIM_C * DIM_C * 2;     //  2 MB each
  _Float16* xh  = (_Float16*)(ws);
  _Float16* Wqh = (_Float16*)(ws + SZ_X);
  _Float16* Wkh = (_Float16*)(ws + SZ_X + SZ_W);
  _Float16* Wvh = (_Float16*)(ws + SZ_X + 2 * SZ_W);
  _Float16* Woh = (_Float16*)(ws + SZ_X + 3 * SZ_W);
  _Float16* Qh  = (_Float16*)(ws + SZ_X + 4 * SZ_W);
  _Float16* Kh  = (_Float16*)(ws + 2 * SZ_X + 4 * SZ_W);
  _Float16* Vth = (_Float16*)(ws + 3 * SZ_X + 4 * SZ_W);
  _Float16* AOh = (_Float16*)(ws + 4 * SZ_X + 4 * SZ_W);
  unsigned int* pm = (unsigned int*)(ws + 5 * SZ_X + 4 * SZ_W);

  cvt_f2h<<<4096, 256, 0, stream>>>(x,  xh,  BN * DIM_C);
  cvt_f2h<<<2048, 256, 0, stream>>>(Wq, Wqh, DIM_C * DIM_C);
  cvt_f2h<<<2048, 256, 0, stream>>>(Wk, Wkh, DIM_C * DIM_C);
  cvt_f2h<<<2048, 256, 0, stream>>>(Wv, Wvh, DIM_C * DIM_C);
  cvt_f2h<<<2048, 256, 0, stream>>>(Wo, Woh, DIM_C * DIM_C);

  const int pm_total = BATCH * N_SEQ * (N_SEQ / 32);
  pack_mask<<<1024, 256, 0, stream>>>(mask, pm, pm_total);

  {
    dim3 grid(BN / 64, DIM_C / 64, 3);
    gemm_qkv<<<grid, 128, 0, stream>>>(xh, Wqh, Wkh, Wvh, bq, bk, bv,
                                       Qh, Kh, Vth);
  }
  {
    dim3 grid(N_SEQ / 32, NUM_HEADS, BATCH);
    attention<<<grid, 32, 0, stream>>>(Qh, Kh, Vth, pm, AOh);
  }
  {
    dim3 grid(BN / 64, DIM_C / 64);
    gemm_out<<<grid, 128, 0, stream>>>(AOh, Woh, bo, out);
  }

  (void)in_sizes; (void)n_in; (void)out_size; (void)ws_size;
}
